// MultiHeadAttention_15968688407015
// MI455X (gfx1250) — matrix operand order repaired and checked by reading, not compiled, not run on hardware
//
#include <hip/hip_runtime.h>
#include <hip/hip_bf16.h>

typedef __bf16 bf16_t;
typedef __attribute__((ext_vector_type(16))) __bf16 v16bf;
typedef __attribute__((ext_vector_type(8)))  __bf16 v8bf;
typedef __attribute__((ext_vector_type(8)))  float  v8f;
typedef __attribute__((ext_vector_type(4)))  float  v4f;

#define DIMX   768
#define NHEADS 12
#define HDIM   64
#define NB     2
#define NSEQ   2048
#define MROWS  (NB*NSEQ)   // 4096

__device__ __forceinline__ v16bf cat8(v8bf a, v8bf b) {
  return __builtin_shufflevector(a, b, 0,1,2,3,4,5,6,7,8,9,10,11,12,13,14,15);
}

__device__ __forceinline__ v8f vexp8(v8f s) {
  v8f r;
#pragma unroll
  for (int i = 0; i < 8; ++i) r[i] = __expf(s[i]);
  return r;
}

// ---------------------------------------------------------------------------
// Kernel 1: fp32 -> bf16 conversion
// ---------------------------------------------------------------------------
__global__ void cvt_f32_bf16_kernel(const float* __restrict__ src,
                                    bf16_t* __restrict__ dst, int n) {
  int i = blockIdx.x * blockDim.x + threadIdx.x;
  if (i < n) dst[i] = (bf16_t)src[i];
}

// ---------------------------------------------------------------------------
// Kernel 2: QKV projection  y = x @ W^T  (M=4096, N=768, K=768), bf16 WMMA.
// blockIdx.y selects matrix (0=Q,1=K,2=V). 4 waves/block, one 16x16 tile/wave.
// Q,K stored as [(b*12+h)][n][d] row-major (A/B friendly for scores).
// V stored transposed: Vt[(b*12+h)][d][k]  (B-operand friendly for attn@V).
// ---------------------------------------------------------------------------
__global__ __launch_bounds__(128) void qkv_proj_kernel(
    const bf16_t* __restrict__ xb,
    const bf16_t* __restrict__ Wqb,
    const bf16_t* __restrict__ Wkb,
    const bf16_t* __restrict__ Wvb,
    bf16_t* __restrict__ Qb,
    bf16_t* __restrict__ Kb,
    bf16_t* __restrict__ Vt) {
  const int lane = threadIdx.x & 31;
  const int wv   = threadIdx.x >> 5;
  const int mat  = blockIdx.y;                 // 0=Q 1=K 2=V
  const int tile = blockIdx.x * 4 + wv;        // 0..12287
  const int tn   = tile % (DIMX / 16);         // col tile 0..47
  const int tm   = tile / (DIMX / 16);         // row tile 0..255
  const int m0   = tm * 16;
  const int c0   = tn * 16;
  const bf16_t* W = (mat == 0) ? Wqb : ((mat == 1) ? Wkb : Wvb);

  const int lm  = lane & 15;
  const int lhi = lane >> 4;
  const int rowA  = m0 + lm;        // A: lane holds row m = lane%16
  const int kselA = lhi * 8;        // A half-offset per lane group
  const int rowB  = c0 + lm;        // B: W row n
  const int kselB = lhi * 8;        // B K-offset per lane group

  v8f acc = {};
  for (int kb = 0; kb < DIMX; kb += 32) {
    v8bf a0 = *(const v8bf*)(xb + (size_t)rowA * DIMX + kb + kselA);
    v8bf a1 = *(const v8bf*)(xb + (size_t)rowA * DIMX + kb + 16 + kselA);
    v16bf A = cat8(a0, a1);
    v8bf b0 = *(const v8bf*)(W + (size_t)rowB * DIMX + kb + kselB);
    v8bf b1 = *(const v8bf*)(W + (size_t)rowB * DIMX + kb + 16 + kselB);
    v16bf Bm = cat8(b0, b1);
    acc = __builtin_amdgcn_wmma_f32_16x16x32_bf16(false, A, false, Bm,
                                                  (short)0, acc, false, false);
  }

  // D tile: element j <-> m_local = lhi*8 + j, n_local = lm
  const int h    = c0 / HDIM;
  const int ds   = c0 % HDIM;
  const int bidx = m0 / NSEQ;
  const int q0   = (m0 % NSEQ) + lhi * 8;
  if (mat == 2) {
    // Vt[(b*12+h)*64 + ds + lm][k = q0..q0+7]: contiguous b128 store
    v8bf pk;
#pragma unroll
    for (int j = 0; j < 8; ++j) pk[j] = (bf16_t)acc[j];
    *(v8bf*)(Vt + ((size_t)((bidx * NHEADS + h) * HDIM + ds + lm)) * NSEQ + q0) = pk;
  } else {
    bf16_t* dst = (mat == 0) ? Qb : Kb;
    bf16_t* base = dst + ((size_t)(bidx * NHEADS + h)) * NSEQ * HDIM;
#pragma unroll
    for (int j = 0; j < 8; ++j)
      base[(size_t)(q0 + j) * HDIM + ds + lm] = (bf16_t)acc[j];
  }
}

// ---------------------------------------------------------------------------
// Kernel 3: fused scores + head-softmax + attn@V.
// Block = 4 waves, owns (batch b, 16-query tile); wave w handles heads 3w..3w+2.
// Streams key chunks of 32. Per chunk: 12 score WMMA + 12 out WMMA per wave,
// exp in f32, cross-wave LDS reduction of sum over the 12 heads -> Z[q,k].
// attn = exp(s) * rcp(Z) / sqrt(768), bounced through LDS into bf16 A-layout.
// ---------------------------------------------------------------------------
__global__ __launch_bounds__(128) void attn_head_softmax_kernel(
    const bf16_t* __restrict__ Qb,
    const bf16_t* __restrict__ Kb,
    const bf16_t* __restrict__ Vt,
    float* __restrict__ out) {
  __shared__ float zpart[2][4][256];   // [ktile][wave][k*16+q] partial head-sums
  __shared__ float abuf[4][32][16];    // per-wave attn bounce: [k][q]

  const int lane = threadIdx.x & 31;
  const int wv   = threadIdx.x >> 5;
  const int bidx = blockIdx.x >> 7;            // /128
  const int q0   = (blockIdx.x & 127) * 16;
  const int h0   = wv * 3;
  const int lm   = lane & 15;
  const int lhi  = lane >> 4;

  // Resident Q A-tiles: 3 heads x 2 halves of d (each 16x32 bf16)
  v16bf QA[3][2];
#pragma unroll
  for (int hh = 0; hh < 3; ++hh) {
    const bf16_t* qrow =
        Qb + ((size_t)((bidx * NHEADS + h0 + hh) * NSEQ + q0 + lm)) * HDIM;
#pragma unroll
    for (int half = 0; half < 2; ++half) {
      v8bf a0 = *(const v8bf*)(qrow + half * 32 + lhi * 8);
      v8bf a1 = *(const v8bf*)(qrow + half * 32 + 16 + lhi * 8);
      QA[hh][half] = cat8(a0, a1);
    }
  }

  v8f outAcc[3][4];
#pragma unroll
  for (int hh = 0; hh < 3; ++hh)
#pragma unroll
    for (int dt = 0; dt < 4; ++dt) outAcc[hh][dt] = (v8f){};

  const float inv_s = 0.03608439182435161f;  // 1/sqrt(768)

  for (int kc = 0; kc < NSEQ; kc += 32) {
    // ---- scores + exp for 3 heads, two 16-key tiles ----
    v8f e[3][2];
    v8f psum[2];
#pragma unroll
    for (int kt = 0; kt < 2; ++kt) {
#pragma unroll
      for (int hh = 0; hh < 3; ++hh) {
        const bf16_t* krow =
            Kb + ((size_t)((bidx * NHEADS + h0 + hh) * NSEQ + kc + kt * 16 + lm)) * HDIM;
        v8f s = {};
#pragma unroll
        for (int half = 0; half < 2; ++half) {
          v8bf b0 = *(const v8bf*)(krow + half * 32 + lhi * 8);
          v8bf b1 = *(const v8bf*)(krow + half * 32 + 16 + lhi * 8);
          v16bf Bm = cat8(b0, b1);
          s = __builtin_amdgcn_wmma_f32_16x16x32_bf16(false, QA[hh][half], false,
                                                      Bm, (short)0, s, false, false);
        }
        e[hh][kt] = vexp8(s);  // D tile: element j <-> q=lhi*8+j, k=lm
      }
      psum[kt] = e[0][kt] + e[1][kt] + e[2][kt];
    }

    // ---- cross-wave reduction of sum over heads -> Z[q,k] ----
    __syncthreads();  // prior iteration finished reading zpart
#pragma unroll
    for (int kt = 0; kt < 2; ++kt) {
      float* p = &zpart[kt][wv][lm * 16 + lhi * 8];
      *(v4f*)p       = __builtin_shufflevector(psum[kt], psum[kt], 0, 1, 2, 3);
      *(v4f*)(p + 4) = __builtin_shufflevector(psum[kt], psum[kt], 4, 5, 6, 7);
    }
    __syncthreads();

    v8f R[2];  // scaled reciprocal of Z, same layout as e tiles
#pragma unroll
    for (int kt = 0; kt < 2; ++kt) {
      v4f z0 = {}, z1 = {};
#pragma unroll
      for (int w = 0; w < 4; ++w) {
        const float* p = &zpart[kt][w][lm * 16 + lhi * 8];
        z0 += *(const v4f*)p;
        z1 += *(const v4f*)(p + 4);
      }
#pragma unroll
      for (int i = 0; i < 4; ++i) {
        R[kt][i]     = __builtin_amdgcn_rcpf(z0[i]) * inv_s;
        R[kt][4 + i] = __builtin_amdgcn_rcpf(z1[i]) * inv_s;
      }
    }

    // ---- per head: attn -> bf16 A-layout (LDS bounce) -> out += attn @ V ----
#pragma unroll
    for (int hh = 0; hh < 3; ++hh) {
#pragma unroll
      for (int kt = 0; kt < 2; ++kt) {
        v8f at = e[hh][kt] * R[kt];
        float* p = &abuf[wv][kt * 16 + lm][lhi * 8];
        *(v4f*)p       = __builtin_shufflevector(at, at, 0, 1, 2, 3);
        *(v4f*)(p + 4) = __builtin_shufflevector(at, at, 4, 5, 6, 7);
      }
      v16bf Aat;  // A layout: lane holds q=lm, kk = lhi*8+i and 16+lhi*8+i
#pragma unroll
      for (int i = 0; i < 8; ++i) {
        Aat[i]     = (bf16_t)abuf[wv][lhi * 8 + i][lm];
        Aat[8 + i] = (bf16_t)abuf[wv][16 + lhi * 8 + i][lm];
      }
      const bf16_t* vrow =
          Vt + ((size_t)((bidx * NHEADS + h0 + hh) * HDIM + lm)) * NSEQ + kc + lhi * 8;
#pragma unroll
      for (int dt = 0; dt < 4; ++dt) {
        v8bf b0 = *(const v8bf*)(vrow + (size_t)dt * 16 * NSEQ);
        v8bf b1 = *(const v8bf*)(vrow + (size_t)dt * 16 * NSEQ + 16);
        v16bf Bm = cat8(b0, b1);
        outAcc[hh][dt] = __builtin_amdgcn_wmma_f32_16x16x32_bf16(
            false, Aat, false, Bm, (short)0, outAcc[hh][dt], false, false);
      }
    }
  }

  // ---- store: out[b][q][(h)*64 + dt*16 + lm], element j <-> q=lhi*8+j ----
#pragma unroll
  for (int hh = 0; hh < 3; ++hh)
#pragma unroll
    for (int dt = 0; dt < 4; ++dt)
#pragma unroll
      for (int j = 0; j < 8; ++j)
        out[((size_t)bidx * NSEQ + q0 + lhi * 8 + j) * DIMX +
            (h0 + hh) * HDIM + dt * 16 + lm] = outAcc[hh][dt][j];
}

// ---------------------------------------------------------------------------
extern "C" void kernel_launch(void* const* d_in, const int* in_sizes, int n_in,
                              void* d_out, int out_size, void* d_ws, size_t ws_size,
                              hipStream_t stream) {
  const float* x  = (const float*)d_in[0];
  const float* Wq = (const float*)d_in[1];
  const float* Wk = (const float*)d_in[2];
  const float* Wv = (const float*)d_in[3];

  char* ws = (char*)d_ws;
  const size_t xb_off  = 0;                                   // 4096*768 bf16
  const size_t wq_off  = xb_off + (size_t)MROWS * DIMX * 2;   // 768*768 bf16
  const size_t wk_off  = wq_off + (size_t)DIMX * DIMX * 2;
  const size_t wv_off  = wk_off + (size_t)DIMX * DIMX * 2;
  const size_t q_off   = wv_off + (size_t)DIMX * DIMX * 2;    // 4096*768 bf16
  const size_t k_off   = q_off + (size_t)MROWS * DIMX * 2;
  const size_t v_off   = k_off + (size_t)MROWS * DIMX * 2;

  bf16_t* xb  = (bf16_t*)(ws + xb_off);
  bf16_t* Wqb = (bf16_t*)(ws + wq_off);
  bf16_t* Wkb = (bf16_t*)(ws + wk_off);
  bf16_t* Wvb = (bf16_t*)(ws + wv_off);
  bf16_t* Qb  = (bf16_t*)(ws + q_off);
  bf16_t* Kb  = (bf16_t*)(ws + k_off);
  bf16_t* Vt  = (bf16_t*)(ws + v_off);

  const int nx = MROWS * DIMX;       // 3,145,728
  const int nw = DIMX * DIMX;        // 589,824
  cvt_f32_bf16_kernel<<<(nx + 255) / 256, 256, 0, stream>>>(x, xb, nx);
  cvt_f32_bf16_kernel<<<(nw + 255) / 256, 256, 0, stream>>>(Wq, Wqb, nw);
  cvt_f32_bf16_kernel<<<(nw + 255) / 256, 256, 0, stream>>>(Wk, Wkb, nw);
  cvt_f32_bf16_kernel<<<(nw + 255) / 256, 256, 0, stream>>>(Wv, Wvb, nw);

  // 12288 tiles of 16x16 per matrix, 4 waves/block -> 3072 blocks, y = matrix id
  qkv_proj_kernel<<<dim3(3072, 3, 1), 128, 0, stream>>>(xb, Wqb, Wkb, Wvb,
                                                        Qb, Kb, Vt);

  // One block per (batch, 16-query tile): 2*128 = 256 blocks of 4 waves
  attn_head_softmax_kernel<<<256, 128, 0, stream>>>(Qb, Kb, Vt, (float*)d_out);
}
